// ImageRGB_75952201662701
// MI455X (gfx1250) — compile-verified, matrix operand order checked by reading, not run on hardware
//
#include <hip/hip_runtime.h>
#include <hip/hip_bf16.h>

typedef __attribute__((ext_vector_type(16))) _Float16 v16h;
typedef __attribute__((ext_vector_type(8)))  float    v8f;

// Problem constants (match setup_inputs: B=32, H=W=512, C=3, f32)
constexpr int B   = 32;
constexpr int H   = 512;
constexpr int W   = 512;
constexpr int HW  = H * W;            // 262144
constexpr int TPB = 256;
constexpr int PPT = 8;                // pixels/thread in stage 2
constexpr float CTR = (512 - 1) * 0.5f;  // 255.5 center for both dims

// ---------------------------------------------------------------------------
// helpers
// ---------------------------------------------------------------------------
__device__ __forceinline__ int reflect_idx(int i, int n) {
    int m = i % (2 * n);
    if (m < 0) m += 2 * n;
    return (m < n) ? m : (2 * n - 1 - m);
}

// Bilinear gather with TF 'reflect' fill; optional H/V flip folded into the
// tap indices (flipping the source image == flipping the gather index).
__device__ __forceinline__ void bilerp3(const float* __restrict__ img,
                                        float sy, float sx, int fh, int fv,
                                        float out[3]) {
    float y0f = floorf(sy), x0f = floorf(sx);
    float wy = sy - y0f,    wx = sx - x0f;
    int y0 = (int)y0f, x0 = (int)x0f;
    int y0r = reflect_idx(y0,     H), y1r = reflect_idx(y0 + 1, H);
    int x0r = reflect_idx(x0,     W), x1r = reflect_idx(x0 + 1, W);
    if (fh) { x0r = W - 1 - x0r; x1r = W - 1 - x1r; }
    if (fv) { y0r = H - 1 - y0r; y1r = H - 1 - y1r; }
    const float* p00 = img + (size_t)(y0r * W + x0r) * 3;
    const float* p01 = img + (size_t)(y0r * W + x1r) * 3;
    const float* p10 = img + (size_t)(y1r * W + x0r) * 3;
    const float* p11 = img + (size_t)(y1r * W + x1r) * 3;
    float w00 = (1.f - wy) * (1.f - wx), w01 = (1.f - wy) * wx;
    float w10 = wy * (1.f - wx),         w11 = wy * wx;
#pragma unroll
    for (int c = 0; c < 3; ++c)
        out[c] = p00[c] * w00 + p01[c] * w01 + p10[c] * w10 + p11[c] * w11;
}

// ---------------------------------------------------------------------------
// Stage 1: flip (index remap) + rotation warp -> intermediate in d_ws
//   1024 blocks per image; img derived from blockIdx -> scalar param loads
// ---------------------------------------------------------------------------
__global__ void __launch_bounds__(TPB)
stage1_flip_rotate(const float* __restrict__ in,
                   const unsigned char* __restrict__ flip_h,
                   const unsigned char* __restrict__ flip_v,
                   const float* __restrict__ angle,
                   float* __restrict__ inter) {
    constexpr int BLOCKS_PER_IMG = HW / TPB;           // 1024
    int img = blockIdx.x >> 10;
    int pix = (blockIdx.x & (BLOCKS_PER_IMG - 1)) * TPB + threadIdx.x;
    int y = pix >> 9;           // W == 512
    int x = pix & (W - 1);

    float a  = angle[img];                 // wave-uniform -> scalar
    float cs = cosf(a), sn = sinf(a);
    int fh = flip_h[img] != 0;
    int fv = flip_v[img] != 0;

    float xc = (float)x - CTR, yc = (float)y - CTR;
    float sx = cs * xc - sn * yc + CTR;   // a=cos, b=-sin
    float sy = sn * xc + cs * yc + CTR;   // c=sin, d= cos

    float v[3];
    bilerp3(in + (size_t)img * HW * 3, sy, sx, fh, fv, v);

    float* o = inter + (size_t)img * HW * 3 + (size_t)pix * 3;
    o[0] = v[0]; o[1] = v[1]; o[2] = v[2];
}

// ---------------------------------------------------------------------------
// Stage 2: zoom warp + brightness clip -> d_out; per-lane f32 partial sums,
// then ONE hi/lo f16 split per lane-channel feeding v_wmma_f32_16x16x32_f16
// against an all-ones B (f32 accumulate) as the cross-lane wave reduction.
// ---------------------------------------------------------------------------
__global__ void __launch_bounds__(TPB)
stage2_zoom_bright_sum(const float* __restrict__ inter,
                       const float* __restrict__ zoom,
                       const float* __restrict__ delta,
                       float* __restrict__ out,
                       float* __restrict__ sums) {
    __shared__ float bsum[3];
    if (threadIdx.x < 3) bsum[threadIdx.x] = 0.f;
    __syncthreads();

    constexpr int BLOCKS_PER_IMG = HW / (TPB * PPT);   // 128
    int img   = blockIdx.x >> 7;
    int lpix0 = (blockIdx.x & (BLOCKS_PER_IMG - 1)) * (TPB * PPT) + threadIdx.x;

    float z  = zoom[img];
    float de = delta[img];
    const float* src = inter + (size_t)img * HW * 3;
    float*       dst = out   + (size_t)img * HW * 3;

    // per-lane f32 partial sums (hot loop = pure f32 FMA/add + vmem)
    float sR = 0.f, sG = 0.f, sB = 0.f;

#pragma unroll
    for (int p = 0; p < PPT; ++p) {
        int pix = lpix0 + p * TPB;           // coalesced across lanes
        int y = pix >> 9, x = pix & (W - 1);
        float sx = z * ((float)x - CTR) + CTR;
        float sy = z * ((float)y - CTR) + CTR;
        float v[3];
        bilerp3(src, sy, sx, 0, 0, v);
        float* o = dst + (size_t)pix * 3;
        float r = fminf(fmaxf(v[0] + de, 0.f), 255.f);
        float g = fminf(fmaxf(v[1] + de, 0.f), 255.f);
        float b = fminf(fmaxf(v[2] + de, 0.f), 255.f);
        o[0] = r; o[1] = g; o[2] = b;
        sR += r; sG += g; sB += b;
    }

    // Exact-ish hi/lo split of the partial (<= 2040) so the f16 WMMA keeps
    // ~f32 accuracy: partial = hi + lo with |lo| <= 0.5, lo f16 err ~1e-4.
    v16h ones;
#pragma unroll
    for (int i = 0; i < 16; ++i) ones[i] = (_Float16)1.0f;

    v16h aR = {}, aG = {}, aB = {};
    _Float16 hR = (_Float16)sR; aR[0] = hR; aR[1] = (_Float16)(sR - (float)hR);
    _Float16 hG = (_Float16)sG; aG[0] = hG; aG[1] = (_Float16)(sG - (float)hG);
    _Float16 hB = (_Float16)sB; aB[0] = hB; aB[1] = (_Float16)(sB - (float)hB);

    // D = A*ones : every packed A element lands in exactly one row-sum, so
    // the grand total of D equals the sum over all 32 lanes per channel.
    v8f accR = {}, accG = {}, accB = {};
    accR = __builtin_amdgcn_wmma_f32_16x16x32_f16(false, aR, false, ones, (short)0, accR, false, false);
    accG = __builtin_amdgcn_wmma_f32_16x16x32_f16(false, aG, false, ones, (short)0, accG, false, false);
    accB = __builtin_amdgcn_wmma_f32_16x16x32_f16(false, aB, false, ones, (short)0, accB, false, false);

    // Collapse D: lanes 0-15 hold rowsums M=0..7 in d[0..7], lanes 16-31 hold
    // M=8..15; sum the 8 regs then add the cross-half partner.
    float tR = 0.f, tG = 0.f, tB = 0.f;
#pragma unroll
    for (int j = 0; j < 8; ++j) { tR += accR[j]; tG += accG[j]; tB += accB[j]; }
    tR += __shfl_xor(tR, 16, 32);
    tG += __shfl_xor(tG, 16, 32);
    tB += __shfl_xor(tB, 16, 32);

    if ((threadIdx.x & 31) == 0) {           // one lane per wave -> LDS atomics
        atomicAdd(&bsum[0], tR);
        atomicAdd(&bsum[1], tG);
        atomicAdd(&bsum[2], tB);
    }
    __syncthreads();
    if (threadIdx.x < 3)
        atomicAdd(&sums[img * 3 + threadIdx.x], bsum[threadIdx.x]);
}

// ---------------------------------------------------------------------------
// Stage 3: contrast about per-image/channel mean, in place on d_out (float4)
//   768 blocks per image; img derived from blockIdx -> scalar mean/contrast
// ---------------------------------------------------------------------------
__global__ void __launch_bounds__(TPB)
stage3_contrast(float* __restrict__ out,
                const float* __restrict__ sums,
                const float* __restrict__ contrast) {
    constexpr int ELEMS_PER_BLOCK = TPB * 4;               // 1024
    constexpr int BLOCKS_PER_IMG  = HW * 3 / ELEMS_PER_BLOCK; // 768 (exact)
    int img  = blockIdx.x / BLOCKS_PER_IMG;
    long flat = (long)blockIdx.x * ELEMS_PER_BLOCK + (long)threadIdx.x * 4;

    float f  = contrast[img];                              // wave-uniform
    float m0 = sums[img * 3 + 0] * (1.0f / (float)HW);
    float m1 = sums[img * 3 + 1] * (1.0f / (float)HW);
    float m2 = sums[img * 3 + 2] * (1.0f / (float)HW);

    float4* o4 = reinterpret_cast<float4*>(out);
    float4 v = o4[flat >> 2];
    float r[4] = {v.x, v.y, v.z, v.w};
#pragma unroll
    for (int e = 0; e < 4; ++e) {
        int c = (int)((flat + e) % 3);
        float m = (c == 0) ? m0 : ((c == 1) ? m1 : m2);
        r[e] = fminf(fmaxf((r[e] - m) * f + m, 0.f), 255.f);
    }
    o4[flat >> 2] = make_float4(r[0], r[1], r[2], r[3]);
}

__global__ void zero_sums(float* __restrict__ sums) {
    sums[threadIdx.x] = 0.f;
}

// ---------------------------------------------------------------------------
extern "C" void kernel_launch(void* const* d_in, const int* in_sizes, int n_in,
                              void* d_out, int out_size, void* d_ws, size_t ws_size,
                              hipStream_t stream) {
    (void)in_sizes; (void)n_in; (void)out_size; (void)ws_size;
    const float*         inp      = (const float*)d_in[0];
    const unsigned char* flip_h   = (const unsigned char*)d_in[1];  // jax bool
    const unsigned char* flip_v   = (const unsigned char*)d_in[2];
    const float*         angle    = (const float*)d_in[3];
    const float*         zoom     = (const float*)d_in[4];
    const float*         delta    = (const float*)d_in[5];
    const float*         contrast = (const float*)d_in[6];
    float* out   = (float*)d_out;
    float* inter = (float*)d_ws;                       // B*HW*3 floats (~100.7 MB)
    float* sums  = inter + (size_t)B * HW * 3;         // B*3 floats

    zero_sums<<<1, B * 3, 0, stream>>>(sums);

    stage1_flip_rotate<<<B * HW / TPB, TPB, 0, stream>>>(
        inp, flip_h, flip_v, angle, inter);

    stage2_zoom_bright_sum<<<B * HW / (TPB * PPT), TPB, 0, stream>>>(
        inter, zoom, delta, out, sums);

    stage3_contrast<<<B * HW * 3 / (TPB * 4), TPB, 0, stream>>>(
        out, sums, contrast);
}
